// MultiComponentLocalField_v2_73933567033567
// MI455X (gfx1250) — matrix operand order repaired and checked by reading, not compiled, not run on hardware
//
#include <hip/hip_runtime.h>

typedef __attribute__((ext_vector_type(16))) _Float16 v16h;
typedef __attribute__((ext_vector_type(8)))  _Float16 v8h;
typedef __attribute__((ext_vector_type(8)))  float    v8f;

#define BB 16
#define LL 2048
#define PP 2048
#define CDIM 256        // C*D = 2*128
#define BP 64           // p-rows per block (4 p-blocks of 16)
#define KC 64           // K per staged buffer = 2 WMMA K-steps of 32
#define NCHUNK (LL / KC)
#define NT 8            // n-tiles (of 16 cd) per wave

// ---------------- prep kernel: run once per launch ----------------
// d_ws layout: [0, 128KB)    emb_table converted to f16, row-major [256][256]
//              [128KB, +2KB) per-byte params float2 { -0.5/w^2, sigmoid(amp) }
__global__ __launch_bounds__(256)
void prep_tables(const float* __restrict__ emb,
                 const float* __restrict__ loc,
                 _Float16* __restrict__ embh,
                 float2* __restrict__ prm)
{
    const int i = blockIdx.x * 256 + threadIdx.x;   // 0..65535
    embh[i] = (_Float16)emb[i];
    if (i < 256) {
        const float w = fabsf(loc[i * 3 + 1]) * 0.02f + 1e-5f;
        float2 pr;
        pr.x = -0.5f / (w * w);
        pr.y = 1.0f / (1.0f + __expf(-loc[i * 3 + 2]));
        prm[i] = pr;
    }
}

// ---------------- main kernel ----------------
__global__ __launch_bounds__(256)
void semantic_field_wmma(const int* __restrict__ bytes,
                         const float* __restrict__ pos,
                         const _Float16* __restrict__ embh,
                         const float2* __restrict__ prm,
                         float* __restrict__ out)
{
    // Double-buffered tiles, 2 K-steps per buffer, WMMA fragment layout.
    __shared__ __attribute__((aligned(32))) _Float16 Afrag[2][2][4][32][16];   // 16 KB
    __shared__ __attribute__((aligned(32))) _Float16 Bfrag[2][2][16][32][16];  // 64 KB
    __shared__ float rsc[2][KC];     // -0.5 / width^2 (gathered)
    __shared__ float ampc[2][KC];    // sigmoid amp    (gathered)
    __shared__ int   idxc[2][KC];    // byte index per l
    __shared__ float poss[BP];

    const int t    = threadIdx.x;
    const int lane = t & 31;
    const int wave = t >> 5;
    const int pb   = wave >> 1;      // 0..3 : 16-row p block
    const int ct   = wave & 1;       // 0..1 : 128-col cd block (8 n-tiles)
    const int b    = blockIdx.y;
    const int p0   = blockIdx.x * BP;

    constexpr float step = 1.0f / (float)(LL - 1);

    if (t < BP) poss[t] = pos[b * PP + p0 + t];

    // per-l params for chunk kc -> buffer kc&1 (written 2 chunks ahead)
    auto fillParams = [&](int kc) {
        if (kc < NCHUNK && t < KC) {
            const int buf = kc & 1;
            const int idx = bytes[b * LL + kc * KC + t];
            const float2 pr = prm[idx];
            rsc[buf][t]  = pr.x;
            ampc[buf][t] = pr.y;
            idxc[buf][t] = idx;
        }
    };

    // A (64p x 64l) + B (64l x 256cd) tiles for chunk kc -> buffer kc&1
    auto fillTiles = [&](int kc) {
        if (kc >= NCHUNK) return;
        const int buf = kc & 1;

        // A tile: 4096 f16, 16 per thread = one full fragment row (32B store).
        {
            const int ebase = t * 16;
            const int kstep = ebase >> 11;          // 0..1
            const int rem   = ebase & 2047;
            const int apb   = rem >> 9;             // 0..3
            const int alane = (rem >> 4) & 31;      // fragment lane
            const float xp  = poss[apb * 16 + (alane & 15)];
            v16h pk;
            #pragma unroll
            for (int j = 0; j < 16; ++j) {
                const int r  = j >> 1;
                const int jj = j & 1;
                // A 16x32 f16 fragment: lanes16-31 shift K by 8; VGPR4-7 shift K by 16
                const int kl = ((r & 4) ? 16 : 0) + ((alane >> 4) * 8)
                             + ((r & 3) * 2) + jj;
                const int k  = kstep * 32 + kl;
                const float d = xp - (float)(kc * KC + k) * step;
                pk[j] = (_Float16)(__expf(rsc[buf][k] * d * d) * ampc[buf][k]);
            }
            *reinterpret_cast<v16h*>(&Afrag[buf][kstep][apb][alane][0]) = pk;
        }

        // B tile: gather f16 emb rows as b128 (8 cols/load, L2-hot, coalesced),
        // pack K-pairs into u32 DS stores in fragment layout.
        #pragma unroll
        for (int i = 0; i < 4; ++i) {
            const int item  = t + i * 256;      // 0..1023
            const int oct   = item & 31;        // cd octet (8 consecutive cd)
            const int kp    = (item >> 5) & 15; // K-pair within k-step
            const int kstep = item >> 9;        // 0..1
            const int g     = kp >> 3;          // K half (lanes 0-15 vs 16-31)
            const int sp    = kp & 7;           // slot pair
            const int k0    = kstep * 32 + ((sp >> 2) * 16) + g * 8 + (sp & 3) * 2;
            const v8h r0 = *reinterpret_cast<const v8h*>(
                               embh + idxc[buf][k0]     * CDIM + oct * 8);
            const v8h r1 = *reinterpret_cast<const v8h*>(
                               embh + idxc[buf][k0 + 1] * CDIM + oct * 8);
            const int ntile = oct >> 1;         // (oct*8+j) >> 4, constant over j
            #pragma unroll
            for (int j = 0; j < 8; ++j) {
                const int cd    = oct * 8 + j;
                const int blane = g * 16 + (cd & 15);
                union { _Float16 h[2]; unsigned int u; } pk;
                pk.h[0] = r0[j];
                pk.h[1] = r1[j];
                *reinterpret_cast<unsigned int*>(
                    &Bfrag[buf][kstep][ntile][blane][sp * 2]) = pk.u;
            }
        }
    };

    v8f acc[NT] = {};

    // ---- pipeline prologue ----
    fillParams(0);
    __syncthreads();
    fillParams(1);
    fillTiles(0);
    __syncthreads();

    // ---- steady state: one barrier per chunk; next-chunk fill overlaps WMMA ----
    #pragma unroll 1
    for (int kc = 0; kc < NCHUNK; ++kc) {
        const int buf = kc & 1;

        fillParams(kc + 2);   // -> params[kc&1] (old contents already consumed)
        fillTiles(kc + 1);    // -> tiles[(kc+1)&1], reads params[(kc+1)&1]

        // compute chunk kc: 2 K-steps x 8 WMMAs, 16 independent MMAs per region
        #pragma unroll
        for (int ks = 0; ks < 2; ++ks) {
            const v16h a = *reinterpret_cast<const v16h*>(
                               Afrag[buf][ks][pb][lane]);
            v16h bm[NT];
            #pragma unroll
            for (int q = 0; q < NT; ++q)
                bm[q] = *reinterpret_cast<const v16h*>(
                            Bfrag[buf][ks][ct * NT + q][lane]);
            #pragma unroll
            for (int q = 0; q < NT; ++q)
                acc[q] = __builtin_amdgcn_wmma_f32_16x16x32_f16(
                    false, a, false, bm[q], (short)0, acc[q], false, false);
        }

        __syncthreads();
    }

    // ---- epilogue: f32 accumulators -> out[b][p][cd] ----
    // D layout: VGPR r -> M = r (lanes 0-15) / 8+r (lanes 16-31); N = lane&15
    const int n    = lane & 15;
    const int mrow = (lane >> 4) * 8;
    #pragma unroll
    for (int nt = 0; nt < NT; ++nt) {
        const int cd = (ct * NT + nt) * 16 + n;
        #pragma unroll
        for (int r = 0; r < 8; ++r) {
            const int p = p0 + pb * 16 + mrow + r;
            out[(size_t)(b * PP + p) * CDIM + cd] = acc[nt][r];
        }
    }
}

extern "C" void kernel_launch(void* const* d_in, const int* in_sizes, int n_in,
                              void* d_out, int out_size, void* d_ws, size_t ws_size,
                              hipStream_t stream) {
    (void)in_sizes; (void)n_in; (void)ws_size; (void)out_size;
    const int*   bytes = (const int*)d_in[0];    // byte_sequence [B,L] int32
    const float* pos   = (const float*)d_in[1];  // continuous_positions [B,P] f32
    const float* emb   = (const float*)d_in[2];  // emb_table [256,256] f32
    const float* loc   = (const float*)d_in[3];  // locality_shapes [256,3] f32
    float* out = (float*)d_out;                  // [B,P,C,D] f32

    _Float16* embh = (_Float16*)d_ws;                          // 128 KB
    float2*   prm  = (float2*)((char*)d_ws + CDIM * CDIM * 2); //   2 KB

    prep_tables<<<dim3(CDIM), 256, 0, stream>>>(emb, loc, embh, prm);

    dim3 grid(PP / BP, BB);
    semantic_field_wmma<<<grid, 256, 0, stream>>>(bytes, pos, embh, prm, out);
}
